// MultiQueryAttention_3599182594591
// MI455X (gfx1250) — matrix operand order repaired and checked by reading, not compiled, not run on hardware
//
#include <hip/hip_runtime.h>
#include <hip/hip_bf16.h>

typedef __attribute__((ext_vector_type(16))) _Float16 v16h;
typedef __attribute__((ext_vector_type(8)))  _Float16 v8h;
typedef __attribute__((ext_vector_type(8)))  float    v8f;

#define WMMA_F16(A, B, C) \
  __builtin_amdgcn_wmma_f32_16x16x32_f16(false, (A), false, (B), (short)0, (C), false, false)

#if defined(__has_builtin)
#if __has_builtin(__builtin_amdgcn_tensor_load_to_lds) && __has_builtin(__builtin_amdgcn_s_wait_tensorcnt)
#define USE_TDM 1
#endif
#endif
#ifndef USE_TDM
#define USE_TDM 0
#endif

// ---- fragment loaders (ISA 7.12.2 layouts) -------------------------------
__device__ __forceinline__ v16h make16(const _Float16* p0, const _Float16* p1) {
  v8h a = *(const v8h*)p0;
  v8h b = *(const v8h*)p1;
  v16h r;
#pragma unroll
  for (int i = 0; i < 8; ++i) { r[i] = a[i]; r[i + 8] = b[i]; }
  return r;
}
// A fragment: lane holds row m=lane%16; K chunks [8h,8h+8) and [16+8h,16+8h+8)
__device__ __forceinline__ v16h load_a_frag(const _Float16* row, int lh) {
  return make16(row + 8 * lh, row + 16 + 8 * lh);
}
// B fragment: lane holds col n=lane%16; K chunks [8h,8h+8) and [16+8h,16+8h+8)
__device__ __forceinline__ v16h load_b_frag(const _Float16* col, int lh) {
  return make16(col + 8 * lh, col + 16 + 8 * lh);
}

// ---- 16-lane butterfly reductions via DPP16 (VALU only, no LDS traffic) --
// xor1 = quad_perm(1,0,3,2)=0xB1, xor2 = quad_perm(2,3,0,1)=0x4E,
// ROW_HALF_MIRROR (0x141) pairs the two quads of each 8-group,
// ROW_MIRROR (0x140) pairs the two 8-groups of each 16-lane row.
template <int CTRL>
__device__ __forceinline__ float dppf(float v) {
  return __int_as_float(
      __builtin_amdgcn_update_dpp(0, __float_as_int(v), CTRL, 0xf, 0xf, true));
}
__device__ __forceinline__ float rmax16(float v) {
  v = fmaxf(v, dppf<0xB1>(v));
  v = fmaxf(v, dppf<0x4E>(v));
  v = fmaxf(v, dppf<0x141>(v));
  v = fmaxf(v, dppf<0x140>(v));
  return v;
}
__device__ __forceinline__ float rsum16(float v) {
  v += dppf<0xB1>(v);
  v += dppf<0x4E>(v);
  v += dppf<0x141>(v);
  v += dppf<0x140>(v);
  return v;
}

#if USE_TDM
// ---- Tensor Data Mover: 64x64 f16 tile, 2-D tensor descriptor (ISA ch.8) --
typedef __attribute__((ext_vector_type(4))) unsigned int u32x4;
typedef __attribute__((ext_vector_type(8))) int i32x8;
typedef __attribute__((ext_vector_type(4))) int i32x4;

__device__ __forceinline__ void tdm_load_tile_f16(unsigned lds_off, const void* gsrc,
                                                  unsigned dim0, unsigned dim1,
                                                  unsigned stride0) {
  const unsigned TILE0 = 64u, TILE1 = 64u;
  unsigned long long ga = (unsigned long long)(uintptr_t)gsrc;
  u32x4 g0;
  g0[0] = 1u;                                            // count=1, user mode
  g0[1] = lds_off;                                       // lds_addr (bytes)
  g0[2] = (unsigned)ga;                                  // global_addr[31:0]
  g0[3] = (unsigned)((ga >> 32) & 0x01ffffffu) | (2u << 30); // addr[56:32] | type=2
  i32x8 g1;
  g1[0] = (int)(1u << 16);                               // wg_mask=0, data_size=1 (2B)
  g1[1] = (int)((dim0 & 0xffffu) << 16);                 // tensor_dim0[15:0]
  g1[2] = (int)((dim0 >> 16) | ((dim1 & 0xffffu) << 16));// dim0 hi | dim1 lo
  g1[3] = (int)((dim1 >> 16) | (TILE0 << 16));           // dim1 hi | tile_dim0
  g1[4] = (int)TILE1;                                    // tile_dim1, tile_dim2=0
  g1[5] = (int)stride0;                                  // tensor_dim0_stride[31:0]
  g1[6] = 0;                                             // stride0 hi, stride1 lo
  g1[7] = 0;
  i32x4 z4 = (i32x4)(0);
  i32x8 z8 = (i32x8)(0);
  __builtin_amdgcn_tensor_load_to_lds(g0, g1, z4, z4, z8, 0);
}
#endif

// ---- elementwise prep ----------------------------------------------------
__global__ void cvt_f16_kernel(const float* __restrict__ x, _Float16* __restrict__ y, int n) {
  int i = blockIdx.x * blockDim.x + threadIdx.x;
  if (i < n) y[i] = (_Float16)x[i];
}

// Wt[n*K + k] = (f16)W[k*N + n]   (W stored [in=K, out=N])
__global__ void wtrans_kernel(const float* __restrict__ w, _Float16* __restrict__ wt, int K, int N) {
  int i = blockIdx.x * blockDim.x + threadIdx.x;
  if (i >= K * N) return;
  int n = i % N, k = i / N;
  wt[(size_t)n * K + k] = (_Float16)w[i];
}

// ---- tiled WMMA GEMM: C[.,N](f32) = A[.,K](f16,rm) * Bt[N,K](f16,rm)^T ---
template <int N, int K>
__launch_bounds__(256)
__global__ void gemm_f16_kernel(const _Float16* __restrict__ A, const _Float16* __restrict__ Bt,
                                float* __restrict__ C) {
  __shared__ __align__(16) _Float16 As[128 * 32];
  __shared__ __align__(16) _Float16 Bs[128 * 32];
  const int tid  = threadIdx.x;
  const int wave = tid >> 5, lane = tid & 31;
  const int wm = wave >> 1, wn = wave & 1;      // 4 (M) x 2 (N) waves
  const int ln = lane & 15, lh = lane >> 4;
  const int rowblk = blockIdx.y * 128, colblk = blockIdx.x * 128;

  // staging: 512 16B chunks per operand, 2 per thread (rows r0 and r0+64)
  const int r0 = tid >> 2, kc0 = (tid & 3) * 8;
  const _Float16* aP0 = A + (size_t)(rowblk + r0) * K + kc0;
  const _Float16* aP1 = aP0 + (size_t)64 * K;
  const _Float16* bP0 = Bt + (size_t)(colblk + r0) * K + kc0;
  const _Float16* bP1 = bP0 + (size_t)64 * K;

  v8h ra0 = *(const v8h*)aP0, ra1 = *(const v8h*)aP1;
  v8h rb0 = *(const v8h*)bP0, rb1 = *(const v8h*)bP1;

  v8f acc[2][4];
#pragma unroll
  for (int mt = 0; mt < 2; ++mt)
#pragma unroll
    for (int nt = 0; nt < 4; ++nt) acc[mt][nt] = (v8f)(0.0f);

  for (int k0 = 0; k0 < K; k0 += 32) {
    *(v8h*)&As[r0 * 32 + kc0]        = ra0;
    *(v8h*)&As[(r0 + 64) * 32 + kc0] = ra1;
    *(v8h*)&Bs[r0 * 32 + kc0]        = rb0;
    *(v8h*)&Bs[(r0 + 64) * 32 + kc0] = rb1;
    __syncthreads();

    if (k0 + 32 < K) {                       // prefetch next K-tile (overlaps WMMA)
      aP0 += 32; aP1 += 32; bP0 += 32; bP1 += 32;
      ra0 = *(const v8h*)aP0; ra1 = *(const v8h*)aP1;
      rb0 = *(const v8h*)bP0; rb1 = *(const v8h*)bP1;
    }

    v16h afr[2], bfr[4];
#pragma unroll
    for (int mt = 0; mt < 2; ++mt)
      afr[mt] = load_a_frag(&As[(wm * 32 + mt * 16 + ln) * 32], lh);
#pragma unroll
    for (int nt = 0; nt < 4; ++nt)
      bfr[nt] = load_b_frag(&Bs[(wn * 64 + nt * 16 + ln) * 32], lh);
#pragma unroll
    for (int mt = 0; mt < 2; ++mt)
#pragma unroll
      for (int nt = 0; nt < 4; ++nt)
        acc[mt][nt] = WMMA_F16(afr[mt], bfr[nt], acc[mt][nt]);
    __syncthreads();
  }

#pragma unroll
  for (int mt = 0; mt < 2; ++mt)
#pragma unroll
    for (int nt = 0; nt < 4; ++nt) {
      float* cp = C + (size_t)(rowblk + wm * 32 + mt * 16 + 8 * lh) * N
                    + colblk + wn * 64 + nt * 16 + ln;
#pragma unroll
      for (int r = 0; r < 8; ++r) cp[(size_t)r * N] = acc[mt][nt][r];
    }
}

// ---- RoPE + layout shuffles ---------------------------------------------
__global__ void rope_q_kernel(const float* __restrict__ Qp, _Float16* __restrict__ Qh, int L) {
  int idx = blockIdx.x * blockDim.x + threadIdx.x;       // L*16*32
  if (idx >= L * 16 * 32) return;
  int i = idx & 31, t = idx >> 5;
  int h = t & 15, l = t >> 4;
  float invf = __powf(10000.0f, -(float)i / 32.0f);
  float ang = (float)l * invf;
  float c = __cosf(ang), s = __sinf(ang);
  const float* src = Qp + (size_t)l * 1024 + h * 64;
  float x1 = src[i], x2 = src[i + 32];
  _Float16* dst = Qh + ((size_t)h * L + l) * 64;
  const float sc = 0.125f;                               // fold 1/sqrt(dk)
  dst[i]      = (_Float16)((x1 * c - x2 * s) * sc);
  dst[i + 32] = (_Float16)((x2 * c + x1 * s) * sc);
}

__global__ void rope_k_kernel(const float* __restrict__ Kp, _Float16* __restrict__ Kh, int L) {
  int idx = blockIdx.x * blockDim.x + threadIdx.x;       // L*2*32
  if (idx >= L * 2 * 32) return;
  int i = idx & 31, t = idx >> 5;
  int kv = t & 1, l = t >> 1;
  float invf = __powf(10000.0f, -(float)i / 32.0f);
  float ang = (float)l * invf;
  float c = __cosf(ang), s = __sinf(ang);
  const float* src = Kp + (size_t)l * 128 + kv * 64;
  float x1 = src[i], x2 = src[i + 32];
  _Float16* dst = Kh + ((size_t)kv * L + l) * 64;
  dst[i]      = (_Float16)(x1 * c - x2 * s);
  dst[i + 32] = (_Float16)(x2 * c + x1 * s);
}

__global__ void vtrans_kernel(const float* __restrict__ Vp, _Float16* __restrict__ Vt, int L) {
  int idx = blockIdx.x * blockDim.x + threadIdx.x;       // L*128
  if (idx >= L * 128) return;
  int d = idx & 127, l = idx >> 7;
  int kv = d >> 6, dk = d & 63;
  Vt[((size_t)kv * 64 + dk) * L + l] = (_Float16)Vp[idx];
}

// ---- flash attention (causal, MQA) --------------------------------------
// grid (L/64, H); 128 threads = 4 waves; wave w owns 16 query rows.
// K/V tiles staged by the Tensor Data Mover (double-buffered) when available.
__launch_bounds__(128)
__global__ void mqa_attn_kernel(const _Float16* __restrict__ Qh,  // [16][L][64]
                                const _Float16* __restrict__ Kh,  // [2][L][64]
                                const _Float16* __restrict__ Vt,  // [2][64][L]
                                _Float16* __restrict__ AO,        // [L][1024]
                                int L) {
#if USE_TDM
  __shared__ __align__(16) _Float16 Ks[2][64 * 64];    // [key][dk]
  __shared__ __align__(16) _Float16 Vs[2][64 * 64];    // [dk][key]
#else
  __shared__ __align__(16) _Float16 Ks[1][64 * 64];
  __shared__ __align__(16) _Float16 Vs[1][64 * 64];
#endif
  __shared__ __align__(16) _Float16 Pb[4][16 * 64];    // per-wave P transpose buffer

  const int h = blockIdx.y, kvh = h >> 3;
  const int qb = blockIdx.x;
  const int tid = threadIdx.x, wave = tid >> 5, lane = tid & 31;
  const int ln = lane & 15, lh = lane >> 4;
  const int qrow0 = qb * 64 + wave * 16;

  const _Float16* kbase = Kh + (size_t)kvh * L * 64;   // [L][64] rm
  const _Float16* vbase = Vt + (size_t)kvh * 64 * L;   // [64][L] rm

  // Q fragments, resident all kernel (already scaled by 1/sqrt(dk))
  const _Float16* qptr = Qh + ((size_t)h * L + qrow0 + ln) * 64;
  v16h qf0 = load_a_frag(qptr, lh);
  v16h qf1 = load_a_frag(qptr + 32, lh);

  v8f o[4];
#pragma unroll
  for (int nt = 0; nt < 4; ++nt) o[nt] = (v8f)(0.0f);
  float mrow[8], lrow[8];
#pragma unroll
  for (int r = 0; r < 8; ++r) { mrow[r] = -1e30f; lrow[r] = 0.0f; }

#if USE_TDM
  if (wave == 0) {                                     // kick off block 0 DMA
    tdm_load_tile_f16((unsigned)(uintptr_t)&Ks[0][0], kbase, 64u, (unsigned)L, 64u);
    tdm_load_tile_f16((unsigned)(uintptr_t)&Vs[0][0], vbase, (unsigned)L, 64u, (unsigned)L);
  }
#endif

  for (int jb = 0; jb <= qb; ++jb) {
#if USE_TDM
    if (wave == 0) {
      if (jb < qb) {                                   // DMA next block, then wait
        int nb = (jb + 1) & 1;                         // for current (<=2 in flight)
        tdm_load_tile_f16((unsigned)(uintptr_t)&Ks[nb][0],
                          kbase + (size_t)(jb + 1) * 64 * 64, 64u, (unsigned)L, 64u);
        tdm_load_tile_f16((unsigned)(uintptr_t)&Vs[nb][0],
                          vbase + (size_t)(jb + 1) * 64, (unsigned)L, 64u, (unsigned)L);
        __builtin_amdgcn_s_wait_tensorcnt(2);
      } else {
        __builtin_amdgcn_s_wait_tensorcnt(0);
      }
    }
    __syncthreads();
    const _Float16* KsB = &Ks[jb & 1][0];
    const _Float16* VsB = &Vs[jb & 1][0];
#else
#pragma unroll
    for (int i = 0; i < 4; ++i) {
      int c = tid + i * 128;
      int r = c >> 3, kc = (c & 7) * 8;
      *(v8h*)&Ks[0][r * 64 + kc] = *(const v8h*)&kbase[((size_t)jb * 64 + r) * 64 + kc];
      *(v8h*)&Vs[0][r * 64 + kc] = *(const v8h*)&vbase[(size_t)r * L + jb * 64 + kc];
    }
    __syncthreads();
    const _Float16* KsB = &Ks[0][0];
    const _Float16* VsB = &Vs[0][0];
#endif

    // S(16x64) = Q * K^T
    v8f s[4];
#pragma unroll
    for (int nt = 0; nt < 4; ++nt) {
      const _Float16* kp = &KsB[(nt * 16 + ln) * 64];
      v16h b0 = load_b_frag(kp, lh);
      v16h b1 = load_b_frag(kp + 32, lh);
      v8f z = (v8f)(0.0f);
      z = WMMA_F16(qf0, b0, z);
      z = WMMA_F16(qf1, b1, z);
      s[nt] = z;
    }

    if (jb == qb) {                            // causal mask on diagonal block
#pragma unroll
      for (int nt = 0; nt < 4; ++nt)
#pragma unroll
        for (int r = 0; r < 8; ++r) {
          int key = jb * 64 + nt * 16 + ln;
          int row = qrow0 + r + 8 * lh;
          if (key > row) s[nt][r] = -1e30f;
        }
    }

    // online softmax (rows in regs r, keys across 16-lane halves)
    float alpha[8];
#pragma unroll
    for (int r = 0; r < 8; ++r) {
      float ml = fmaxf(fmaxf(s[0][r], s[1][r]), fmaxf(s[2][r], s[3][r]));
      ml = rmax16(ml);
      float mn = fmaxf(mrow[r], ml);
      alpha[r] = __expf(mrow[r] - mn);
      mrow[r] = mn;
      lrow[r] *= alpha[r];
    }
#pragma unroll
    for (int nt = 0; nt < 4; ++nt)
#pragma unroll
      for (int r = 0; r < 8; ++r) o[nt][r] *= alpha[r];

    float rs[8];
#pragma unroll
    for (int r = 0; r < 8; ++r) rs[r] = 0.0f;
#pragma unroll
    for (int nt = 0; nt < 4; ++nt)
#pragma unroll
      for (int r = 0; r < 8; ++r) {
        float p = __expf(s[nt][r] - mrow[r]);
        rs[r] += p;
        Pb[wave][(r + 8 * lh) * 64 + nt * 16 + ln] = (_Float16)p;  // C->A transpose
      }
#pragma unroll
    for (int r = 0; r < 8; ++r) lrow[r] += rsum16(rs[r]);

    // O(16x64) += P(16x64) * V(64x64)
    v16h p0 = load_a_frag(&Pb[wave][ln * 64], lh);
    v16h p1 = load_a_frag(&Pb[wave][ln * 64 + 32], lh);
#pragma unroll
    for (int nt = 0; nt < 4; ++nt) {
      const _Float16* vp = &VsB[(nt * 16 + ln) * 64];
      v16h v0 = load_b_frag(vp, lh);
      v16h v1 = load_b_frag(vp + 32, lh);
      o[nt] = WMMA_F16(p0, v0, o[nt]);
      o[nt] = WMMA_F16(p1, v1, o[nt]);
    }
    __syncthreads();
  }

#pragma unroll
  for (int nt = 0; nt < 4; ++nt)
#pragma unroll
    for (int r = 0; r < 8; ++r) {
      int row = qrow0 + r + 8 * lh;
      int col = h * 64 + nt * 16 + ln;
      AO[(size_t)row * 1024 + col] = (_Float16)(o[nt][r] / lrow[r]);
    }
}

// ---- host-side orchestration --------------------------------------------
extern "C" void kernel_launch(void* const* d_in, const int* in_sizes, int n_in,
                              void* d_out, int out_size, void* d_ws, size_t ws_size,
                              hipStream_t stream) {
  const int L = 2048, D = 1024, DKV = 128;
  const float* q  = (const float*)d_in[0];
  const float* k  = (const float*)d_in[1];
  const float* v  = (const float*)d_in[2];
  /* d_in[3] = causal mask, computed analytically in-kernel */
  const float* wq = (const float*)d_in[4];
  const float* wk = (const float*)d_in[5];
  const float* wv = (const float*)d_in[6];
  const float* wo = (const float*)d_in[7];
  float* out = (float*)d_out;

  char* ws = (char*)d_ws;
  size_t off = 0;
  auto alloc = [&](size_t bytes) -> void* {
    void* p = ws + off;
    off += (bytes + 255) & ~(size_t)255;
    return p;
  };
  _Float16* qf16 = (_Float16*)alloc((size_t)L * D * 2);
  _Float16* kf16 = (_Float16*)alloc((size_t)L * D * 2);
  _Float16* vf16 = (_Float16*)alloc((size_t)L * D * 2);
  _Float16* wqT  = (_Float16*)alloc((size_t)D * D * 2);
  _Float16* wkT  = (_Float16*)alloc((size_t)D * DKV * 2);
  _Float16* wvT  = (_Float16*)alloc((size_t)D * DKV * 2);
  _Float16* woT  = (_Float16*)alloc((size_t)D * D * 2);
  float*    Qp   = (float*)alloc((size_t)L * D * 4);
  float*    Kp   = (float*)alloc((size_t)L * DKV * 4);
  float*    Vp   = (float*)alloc((size_t)L * DKV * 4);
  _Float16* Qh   = (_Float16*)alloc((size_t)16 * L * 64 * 2);
  _Float16* Kh   = (_Float16*)alloc((size_t)2 * L * 64 * 2);
  _Float16* Vt   = (_Float16*)alloc((size_t)2 * 64 * L * 2);
  _Float16* AO   = (_Float16*)alloc((size_t)L * D * 2);

  const int nLD = L * D;
  cvt_f16_kernel<<<nLD / 256, 256, 0, stream>>>(q, qf16, nLD);
  cvt_f16_kernel<<<nLD / 256, 256, 0, stream>>>(k, kf16, nLD);
  cvt_f16_kernel<<<nLD / 256, 256, 0, stream>>>(v, vf16, nLD);
  wtrans_kernel<<<(D * D) / 256, 256, 0, stream>>>(wq, wqT, D, D);
  wtrans_kernel<<<(D * DKV) / 256, 256, 0, stream>>>(wk, wkT, D, DKV);
  wtrans_kernel<<<(D * DKV) / 256, 256, 0, stream>>>(wv, wvT, D, DKV);
  wtrans_kernel<<<(D * D) / 256, 256, 0, stream>>>(wo, woT, D, D);

  gemm_f16_kernel<1024, 1024><<<dim3(D / 128, L / 128), 256, 0, stream>>>(qf16, wqT, Qp);
  gemm_f16_kernel<128, 1024><<<dim3(DKV / 128, L / 128), 256, 0, stream>>>(kf16, wkT, Kp);
  gemm_f16_kernel<128, 1024><<<dim3(DKV / 128, L / 128), 256, 0, stream>>>(vf16, wvT, Vp);

  rope_q_kernel<<<(L * 16 * 32) / 256, 256, 0, stream>>>(Qp, Qh, L);
  rope_k_kernel<<<(L * 2 * 32) / 256, 256, 0, stream>>>(Kp, Kh, L);
  vtrans_kernel<<<(L * 128) / 256, 256, 0, stream>>>(Vp, Vt, L);

  mqa_attn_kernel<<<dim3(L / 64, 16), 128, 0, stream>>>(Qh, Kh, Vt, AO, L);

  gemm_f16_kernel<1024, 1024><<<dim3(D / 128, L / 128), 256, 0, stream>>>(AO, woT, out);
}
